// SAKELayer_8031588844292
// MI455X (gfx1250) — matrix operand order repaired and checked by reading, not compiled, not run on hardware
//
#include <hip/hip_runtime.h>
#include <hip/hip_bf16.h>

typedef __attribute__((ext_vector_type(16))) _Float16 v16h;
typedef __attribute__((ext_vector_type(2)))  _Float16 v2h;
typedef __attribute__((ext_vector_type(8)))  float    v8f;
typedef __attribute__((ext_vector_type(2)))  float    v2f;

#define N_ATOMS 640
#define NJB     (N_ATOMS/32)   /* 20 j-blocks of 32 */
#define HEADS   4
#define JCHUNK  64
#define NCHUNK  (N_ATOMS / JCHUNK)
#define EPSV    1e-5f
#define INFV    1e5f

union FragU { v16h v; _Float16 h[16]; };
union FragB { v16h v; uint4 q[2]; };
union PackU { uint4 q; _Float16 h[8]; };
union Pair2 { unsigned int u; _Float16 h[2]; v2h v; };

__device__ __forceinline__ float siluf(float v){ return v / (1.0f + __expf(-v)); }
__device__ __forceinline__ float lrelu(float v){ return v > 0.f ? v : 0.2f * v; }

// A-operand fragment: 16x32 f16 matrix (row-major, stride ld), ISA 7.12.2 layout.
__device__ __forceinline__ v16h ldsA(const _Float16* s, int ld){
  int lane = threadIdx.x & 31;
  int m    = lane & 15;
  int koff = (lane >> 4) * 8;
  FragU f;
#pragma unroll
  for (int r = 0; r < 8; ++r){
    int kb = ((r < 4) ? 2*r : 16 + 2*(r - 4)) + koff;
    f.h[2*r]   = s[m*ld + kb];
    f.h[2*r+1] = s[m*ld + kb + 1];
  }
  return f.v;
}

// B-operand fragment: 32x16 (KxN) f16 matrix (row-major, stride ld).
__device__ __forceinline__ v16h ldsB(const _Float16* s, int ld){
  int lane = threadIdx.x & 31;
  int n    = lane & 15;
  int koff = (lane >> 4) * 8;
  FragU f;
#pragma unroll
  for (int r = 0; r < 8; ++r){
    int k = ((r < 4) ? 2*r : 16 + 2*(r - 4)) + koff;
    f.h[2*r]   = s[k*ld + n];
    f.h[2*r+1] = s[(k+1)*ld + n];
  }
  return f.v;
}

__device__ __forceinline__ v8f wmma16(v16h a, v16h b, v8f c){
  return __builtin_amdgcn_wmma_f32_16x16x32_f16(false, a, false, b, (short)0, c, false, false);
}

__device__ __forceinline__ float dot2h(v2h a, v2h b, float c){
#if __has_builtin(__builtin_amdgcn_fdot2)
  return __builtin_amdgcn_fdot2(a, b, c, false);
#else
  return c + (float)a[0]*(float)b[0] + (float)a[1]*(float)b[1];
#endif
}

__device__ __forceinline__ float rmax16(float v){
#pragma unroll
  for (int o = 1; o < 16; o <<= 1) v = fmaxf(v, __shfl_xor(v, o, 32));
  return v;
}
__device__ __forceinline__ float rsum16(float v){
#pragma unroll
  for (int o = 1; o < 16; o <<= 1) v += __shfl_xor(v, o, 32);
  return v;
}
__device__ __forceinline__ void omerge(float& m, float& S, float gm, float gs){
  if (gm > m){ S = S * __expf(m - gm) + gs; m = gm; }
  else       { S += gs * __expf(gm - m); }
}

// ---------------- k0: P = h @ We1[0:64] + be1, Q = h @ We1[64:128] ----------
__global__ void k0_precompute(const float* __restrict__ h,
                              const float* __restrict__ We1,
                              const float* __restrict__ be1,
                              float* __restrict__ P, float* __restrict__ Q){
  __shared__ _Float16 hbuf[16*64];
  __shared__ _Float16 wbuf[32*64];
  int lane = threadIdx.x;
  int it   = blockIdx.x;
  for (int idx = lane; idx < 16*64; idx += 32)
    hbuf[idx] = (_Float16)h[(size_t)(it*16 + (idx>>6))*64 + (idx & 63)];
  int n = lane & 15, hb = lane >> 4;
#pragma unroll
  for (int tgt = 0; tgt < 2; ++tgt){
    int r0 = tgt * 64;
    v8f acc[4] = {};
    for (int kc = 0; kc < 2; ++kc){
      for (int idx = lane; idx < 32*64; idx += 32)
        wbuf[idx] = (_Float16)We1[(size_t)(r0 + kc*32 + (idx>>6))*64 + (idx & 63)];
      v16h a = ldsA(&hbuf[kc*32], 64);
#pragma unroll
      for (int nt = 0; nt < 4; ++nt){
        v16h b = ldsB(&wbuf[nt*16], 64);
        acc[nt] = wmma16(a, b, acc[nt]);
      }
    }
    float* dst = tgt ? Q : P;
#pragma unroll
    for (int nt = 0; nt < 4; ++nt){
      const float* ap = (const float*)&acc[nt];
      float bias = tgt ? 0.f : be1[nt*16 + n];
#pragma unroll
      for (int r = 0; r < 8; ++r){
        int m = r + 8*hb;
        dst[(size_t)(it*16 + m)*64 + nt*16 + n] = ap[r] + bias;
      }
    }
  }
}

// ---------------- k1: edge MLP layer2 via WMMA + logits + online softmax ----
// h_e stored to HBM in fragment-major layout [i][jb][t2][nt][n][j16] so k2 can
// load B fragments as two contiguous b128s.
__global__ void __launch_bounds__(256)
k1_edge(const float* __restrict__ x,
        const float* __restrict__ We1,
        const float* __restrict__ We2, const float* __restrict__ be2,
        const float* __restrict__ Wa,  const float* __restrict__ ba,
        const float* __restrict__ log_gamma,
        const float* __restrict__ P, const float* __restrict__ Q,
        _Float16* __restrict__ he, float* __restrict__ s_out,
        float* __restrict__ part){
  __shared__ float    qbuf[JCHUNK*64];
  __shared__ float    pbuf[16*64];
  __shared__ _Float16 we2buf[64*64];
  __shared__ unsigned int wa2buf[HEADS][32];   // Wa packed as half2 pairs per head
  __shared__ float    w128buf[64], be2buf[64], babuf[HEADS];
  __shared__ float    xi[16*3], xj[JCHUNK*3];
  __shared__ _Float16 tbuf[8][16*64];          // row-major h_e tile (for Wa logit dot)

  int tid  = threadIdx.x;
  int wave = tid >> 5, lane = tid & 31;
  int chunk = blockIdx.x, it = blockIdx.y;
  int j_base = chunk * JCHUNK;

  for (int idx = tid; idx < JCHUNK*64; idx += 256)
    qbuf[idx] = Q[(size_t)(j_base + (idx>>6))*64 + (idx & 63)];
  for (int idx = tid; idx < 16*64; idx += 256)
    pbuf[idx] = P[(size_t)(it*16 + (idx>>6))*64 + (idx & 63)];
  for (int idx = tid; idx < 64*64; idx += 256)
    we2buf[idx] = (_Float16)We2[idx];
  for (int idx = tid; idx < HEADS*32; idx += 256){
    int hd = idx >> 5, kk = idx & 31;
    Pair2 pk;
    pk.h[0] = (_Float16)Wa[(2*kk)*HEADS + hd];
    pk.h[1] = (_Float16)Wa[(2*kk + 1)*HEADS + hd];
    wa2buf[hd][kk] = pk.u;
  }
  for (int idx = tid; idx < 64; idx += 256){
    w128buf[idx] = We1[128*64 + idx]; be2buf[idx] = be2[idx];
  }
  if (tid < HEADS) babuf[tid] = ba[tid];
  for (int idx = tid; idx < 16*3; idx += 256)     xi[idx] = x[(size_t)(it*16)*3 + idx];
  for (int idx = tid; idx < JCHUNK*3; idx += 256) xj[idx] = x[(size_t)j_base*3 + idx];
  __syncthreads();

  int jl = lane & 15, half = lane >> 4;
  int h0 = half * 2;
  float g0  = __expf(log_gamma[h0]), g1 = __expf(log_gamma[h0 + 1]);
  float ba0 = babuf[h0], ba1 = babuf[h0 + 1];

  // loop-invariant We2 B-fragments (kc*4 + nt), kept in registers
  v16h bw[8];
#pragma unroll
  for (int kc = 0; kc < 2; ++kc)
#pragma unroll
    for (int nt = 0; nt < 4; ++nt)
      bw[kc*4 + nt] = ldsB(&we2buf[kc*32*64 + nt*16], 64);

  const v2f* w2 = (const v2f*)&w128buf[0];

  for (int iw = 0; iw < 2; ++iw){
    int iloc = wave*2 + iw;
    int gi   = it*16 + iloc;
    float xix = xi[iloc*3+0], xiy = xi[iloc*3+1], xiz = xi[iloc*3+2];
    float ma0=-3e38f,Sa0=0,mb0=-3e38f,Sb0=0,ms0=-3e38f,Ss0=0;
    float ma1=-3e38f,Sa1=0,mb1=-3e38f,Sb1=0,ms1=-3e38f,Ss1=0;
    const v2f* p2 = (const v2f*)&pbuf[iloc*64];

    for (int jg = 0; jg < JCHUNK/16; ++jg){
      int jq = jg*16 + jl;                    // this lane's pair row (m == jl)
      int gj = j_base + jq;
      float dx0 = xix - xj[jq*3+0], dx1 = xiy - xj[jq*3+1], dx2 = xiz - xj[jq*3+2];
      float nrm = sqrtf(dx0*dx0 + dx1*dx1 + dx2*dx2 + EPSV);

      // A fragments built directly in registers: row m = jl, silu(P+Q+norm*w128)
      // packed-f32 ext-vector math on adjacent k pairs.
      const v2f* q2 = (const v2f*)&qbuf[jq*64];
      FragU fa0, fa1;
      int koff = half * 8;
#pragma unroll
      for (int r = 0; r < 8; ++r){
        int kb  = ((r < 4) ? 2*r : 16 + 2*(r - 4)) + koff;
        int k2i = kb >> 1;
        v2f t0 = p2[k2i]      + q2[k2i]      + nrm * w2[k2i];
        v2f t1 = p2[k2i + 16] + q2[k2i + 16] + nrm * w2[k2i + 16];
        fa0.h[2*r]   = (_Float16)siluf(t0.x);
        fa0.h[2*r+1] = (_Float16)siluf(t0.y);
        fa1.h[2*r]   = (_Float16)siluf(t1.x);
        fa1.h[2*r+1] = (_Float16)siluf(t1.y);
      }

      int tile = chunk*4 + jg;
      int jb = tile >> 1, t2 = tile & 1;
      _Float16* hedst = he + (((size_t)gi*NJB + jb)*2048 + t2*1024);
#pragma unroll
      for (int nt = 0; nt < 4; ++nt){
        v8f acc = {};
        acc = wmma16(fa0.v, bw[nt],     acc);
        acc = wmma16(fa1.v, bw[4 + nt], acc);
        const float* ap = (const float*)&acc;
        PackU pu;
        float bias = be2buf[nt*16 + jl];
#pragma unroll
        for (int r = 0; r < 8; ++r){
          _Float16 hv = (_Float16)siluf(ap[r] + bias);
          pu.h[r] = hv;
          tbuf[wave][(r + 8*half)*64 + nt*16 + jl] = hv;  // row-major copy for logits
        }
        // fragment-major h_e store: [nt][n=jl][j16 = 8*half + r], 16B contiguous
        *(uint4*)(hedst + nt*256 + jl*16 + half*8) = pu.q;
      }

      // logits for this lane's j, two heads: packed f16 dot2 over h_e row
      float b0acc = 0.f, b1acc = 0.f;
      const unsigned int* hrow = (const unsigned int*)&tbuf[wave][jl*64];
      for (int kk = 0; kk < 32; ++kk){
        Pair2 pr; pr.u = hrow[kk];
        Pair2 w0; w0.u = wa2buf[h0][kk];
        Pair2 w1; w1.u = wa2buf[h0 + 1][kk];
        b0acc = dot2h(pr.v, w0.v, b0acc);
        b1acc = dot2h(pr.v, w1.v, b1acc);
      }
      float mask = (gi == gj) ? INFV : 0.f;
      float b0v = lrelu(b0acc + ba0) - mask;
      float b1v = lrelu(b1acc + ba1) - mask;
      float a0v = -(nrm + mask) * g0;
      float a1v = -(nrm + mask) * g1;
      float s0v = a0v + b0v, s1v = a1v + b1v;
      float* sp = s_out + ((size_t)gi*N_ATOMS + gj)*HEADS;
      sp[h0] = s0v; sp[h0 + 1] = s1v;

      float gm, gs;
      gm = rmax16(a0v); gs = rsum16(__expf(a0v - gm)); omerge(ma0, Sa0, gm, gs);
      gm = rmax16(b0v); gs = rsum16(__expf(b0v - gm)); omerge(mb0, Sb0, gm, gs);
      gm = rmax16(s0v); gs = rsum16(__expf(s0v - gm)); omerge(ms0, Ss0, gm, gs);
      gm = rmax16(a1v); gs = rsum16(__expf(a1v - gm)); omerge(ma1, Sa1, gm, gs);
      gm = rmax16(b1v); gs = rsum16(__expf(b1v - gm)); omerge(mb1, Sb1, gm, gs);
      gm = rmax16(s1v); gs = rsum16(__expf(s1v - gm)); omerge(ms1, Ss1, gm, gs);
    }
    if (jl == 0){
      float* p0 = part + (((size_t)gi*HEADS + h0)*NCHUNK + chunk)*6;
      p0[0]=ma0; p0[1]=Sa0; p0[2]=mb0; p0[3]=Sb0; p0[4]=ms0; p0[5]=Ss0;
      float* p1 = part + (((size_t)gi*HEADS + h0 + 1)*NCHUNK + chunk)*6;
      p1[0]=ma1; p1[1]=Sa1; p1[2]=mb1; p1[3]=Sb1; p1[4]=ms1; p1[5]=Ss1;
    }
  }
}

// ---------------- k1b: merge softmax partials -> m, denom ------------------
__global__ void k1b_reduce(const float* __restrict__ part,
                           float* __restrict__ mrow, float* __restrict__ drow){
  int t = blockIdx.x*256 + threadIdx.x;
  if (t >= N_ATOMS*HEADS) return;
  float ma=-3e38f,Sa=0,mb=-3e38f,Sb=0,ms=-3e38f,Ss=0;
  const float* p = part + (size_t)t*NCHUNK*6;
  for (int c = 0; c < NCHUNK; ++c){
    omerge(ma, Sa, p[c*6+0], p[c*6+1]);
    omerge(mb, Sb, p[c*6+2], p[c*6+3]);
    omerge(ms, Ss, p[c*6+4], p[c*6+5]);
  }
  mrow[t] = ms;
  drow[t] = Ss + EPSV * Sa * Sb * __expf(ma + mb - ms);
}

// ---------------- k2: fused attention + spatial aggregation via WMMA --------
// A rows 0-3 = att heads, 4-6 = unit_d, 7-15 = 0; B = h_e fragments straight
// from HBM (fragment-major layout, two b128 loads per fragment).
__global__ void __launch_bounds__(256)
k2_aggregate(const float* __restrict__ x, const _Float16* __restrict__ he,
             const float* __restrict__ s_in,
             const float* __restrict__ mrow, const float* __restrict__ drow,
             float* __restrict__ hagg, float* __restrict__ comb){
  __shared__ _Float16 abuf[8][16*32];
  int tid = threadIdx.x, wave = tid >> 5, lane = tid & 31;
  int i = blockIdx.x*8 + wave;
  float xix = x[(size_t)i*3+0], xiy = x[(size_t)i*3+1], xiz = x[(size_t)i*3+2];
  float m0 = mrow[i*4+0], m1 = mrow[i*4+1], m2 = mrow[i*4+2], m3 = mrow[i*4+3];
  float id0 = 1.f/drow[i*4+0], id1 = 1.f/drow[i*4+1];
  float id2 = 1.f/drow[i*4+2], id3 = 1.f/drow[i*4+3];
  for (int r = 7; r < 16; ++r) abuf[wave][r*32 + lane] = (_Float16)0.f;

  int n = lane & 15, hb = lane >> 4;
  const _Float16* hebase = he + (size_t)i*NJB*2048 + hb*8 + n*16;

  v8f acc[4] = {};
  for (int jb = 0; jb < NJB; ++jb){
    if (jb + 1 < NJB)
      __builtin_prefetch(hebase + (size_t)(jb + 1)*2048, 0, 0);
    // A rows for this lane's j
    int j = jb*32 + lane;
    float dx0 = xix - x[(size_t)j*3+0], dx1 = xiy - x[(size_t)j*3+1], dx2 = xiz - x[(size_t)j*3+2];
    float nrm = sqrtf(dx0*dx0 + dx1*dx1 + dx2*dx2 + EPSV);
    const float* sp = s_in + ((size_t)i*N_ATOMS + j)*HEADS;
    abuf[wave][0*32+lane] = (_Float16)(__expf(sp[0]-m0)*id0);
    abuf[wave][1*32+lane] = (_Float16)(__expf(sp[1]-m1)*id1);
    abuf[wave][2*32+lane] = (_Float16)(__expf(sp[2]-m2)*id2);
    abuf[wave][3*32+lane] = (_Float16)(__expf(sp[3]-m3)*id3);
    float inl = 1.f/(nrm + EPSV);
    abuf[wave][4*32+lane] = (_Float16)(dx0*inl);
    abuf[wave][5*32+lane] = (_Float16)(dx1*inl);
    abuf[wave][6*32+lane] = (_Float16)(dx2*inl);

    v16h a = ldsA(&abuf[wave][0], 32);
    const uint4* g = (const uint4*)(hebase + (size_t)jb*2048);
#pragma unroll
    for (int nt = 0; nt < 4; ++nt){
      FragB u;
      u.q[0] = g[nt*32];
      u.q[1] = g[nt*32 + 128];
      acc[nt] = wmma16(a, u.v, acc[nt]);
    }
  }
  if (lane < 16){
#pragma unroll
    for (int nt = 0; nt < 4; ++nt){
      const float* ap = (const float*)&acc[nt];
      int k = nt*16 + n;
#pragma unroll
      for (int r = 0; r < 4; ++r)
        hagg[(size_t)i*256 + k*4 + r] = ap[r];          // h_e_att: index k*HEADS + head
#pragma unroll
      for (int d = 0; d < 3; ++d)
        comb[((size_t)i*64 + k)*3 + d] = ap[4+d] * (1.0f/N_ATOMS); // comb_sum (mean over j)
    }
  }
}

// helper: 16x64 f16 A @ 64x64 W -> silu(+bias) -> f16 dst (one wave)
__device__ void layer64(const _Float16* A, const float* __restrict__ W,
                        const float* __restrict__ bias,
                        _Float16* wstage, _Float16* dst, int dstld, int dstoff){
  int lane = threadIdx.x & 31;
  int n = lane & 15, hb = lane >> 4;
  v8f acc[4] = {};
  for (int kc = 0; kc < 2; ++kc){
    for (int idx = lane; idx < 32*64; idx += 32)
      wstage[idx] = (_Float16)W[(size_t)(kc*32 + (idx>>6))*64 + (idx & 63)];
    v16h a = ldsA(A + kc*32, 64);
#pragma unroll
    for (int nt = 0; nt < 4; ++nt)
      acc[nt] = wmma16(a, ldsB(&wstage[nt*16], 64), acc[nt]);
  }
#pragma unroll
  for (int nt = 0; nt < 4; ++nt){
    const float* ap = (const float*)&acc[nt];
#pragma unroll
    for (int r = 0; r < 8; ++r){
      int m = r + 8*hb;
      dst[m*dstld + dstoff + nt*16 + n] = (_Float16)siluf(ap[r] + bias[nt*16 + n]);
    }
  }
}

// ---------------- k3: comb_norm, delta_v, post-norm MLP, node MLP ----------
__global__ void k3_node(const float* __restrict__ h, const float* __restrict__ x,
                        const float* __restrict__ Wn1, const float* __restrict__ bn1,
                        const float* __restrict__ Wn2, const float* __restrict__ bn2,
                        const float* __restrict__ Wp1, const float* __restrict__ bp1,
                        const float* __restrict__ Wp2, const float* __restrict__ bp2,
                        const float* __restrict__ Wv,
                        const float* __restrict__ hagg, const float* __restrict__ comb,
                        float* __restrict__ out){
  __shared__ _Float16 feat[16*384];
  __shared__ _Float16 wbuf[32*64];
  __shared__ _Float16 t2[16*64];
  __shared__ _Float16 cnb[16*64];
  int lane = threadIdx.x;
  int it = blockIdx.x, gi0 = it*16;

  for (int idx = lane; idx < 16*64; idx += 32)
    feat[(idx>>6)*384 + (idx & 63)] = (_Float16)h[(size_t)(gi0 + (idx>>6))*64 + (idx & 63)];
  for (int idx = lane; idx < 16*256; idx += 32)
    feat[(idx>>8)*384 + 64 + (idx & 255)] = (_Float16)hagg[(size_t)(gi0 + (idx>>8))*256 + (idx & 255)];
  for (int idx = lane; idx < 16*64; idx += 32){
    int ii = idx >> 6, k = idx & 63;
    const float* c = comb + ((size_t)(gi0 + ii)*64 + k)*3;
    cnb[ii*64 + k] = (_Float16)(c[0]*c[0] + c[1]*c[1] + c[2]*c[2]);
  }
  if (lane < 16){
    int ii = lane;
    float dv0=0.f, dv1=0.f, dv2=0.f;
    for (int k = 0; k < 64; ++k){
      const float* c = comb + ((size_t)(gi0 + ii)*64 + k)*3;
      float wv = Wv[k];
      dv0 += c[0]*wv; dv1 += c[1]*wv; dv2 += c[2]*wv;
    }
    float* xo = out + (size_t)N_ATOMS*64 + (size_t)(gi0 + ii)*3;
    xo[0] = x[(size_t)(gi0 + ii)*3 + 0] + dv0;
    xo[1] = x[(size_t)(gi0 + ii)*3 + 1] + dv1;
    xo[2] = x[(size_t)(gi0 + ii)*3 + 2] + dv2;
  }
  // post-norm MLP: cnb -> Wp1 -> silu -> Wp2 -> silu -> feat[:,320:384]
  layer64(cnb, Wp1, bp1, wbuf, t2, 64, 0);
  layer64(t2,  Wp2, bp2, wbuf, feat, 384, 320);

  // node layer 1: feat (16x384) @ Wn1 (384x64) -> silu -> t2
  int n = lane & 15, hb = lane >> 4;
  {
    v8f acc[4] = {};
    for (int kc = 0; kc < 12; ++kc){
      for (int idx = lane; idx < 32*64; idx += 32)
        wbuf[idx] = (_Float16)Wn1[(size_t)(kc*32 + (idx>>6))*64 + (idx & 63)];
      v16h a = ldsA(&feat[kc*32], 384);
#pragma unroll
      for (int nt = 0; nt < 4; ++nt)
        acc[nt] = wmma16(a, ldsB(&wbuf[nt*16], 64), acc[nt]);
    }
#pragma unroll
    for (int nt = 0; nt < 4; ++nt){
      const float* ap = (const float*)&acc[nt];
#pragma unroll
      for (int r = 0; r < 8; ++r){
        int m = r + 8*hb;
        t2[m*64 + nt*16 + n] = (_Float16)siluf(ap[r] + bn1[nt*16 + n]);
      }
    }
  }
  // node layer 2 + residual -> h_new (f32 out)
  {
    v8f acc[4] = {};
    for (int kc = 0; kc < 2; ++kc){
      for (int idx = lane; idx < 32*64; idx += 32)
        wbuf[idx] = (_Float16)Wn2[(size_t)(kc*32 + (idx>>6))*64 + (idx & 63)];
      v16h a = ldsA(&t2[kc*32], 64);
#pragma unroll
      for (int nt = 0; nt < 4; ++nt)
        acc[nt] = wmma16(a, ldsB(&wbuf[nt*16], 64), acc[nt]);
    }
#pragma unroll
    for (int nt = 0; nt < 4; ++nt){
      const float* ap = (const float*)&acc[nt];
#pragma unroll
      for (int r = 0; r < 8; ++r){
        int m = r + 8*hb;
        int col = nt*16 + n;
        out[(size_t)(gi0 + m)*64 + col] =
            h[(size_t)(gi0 + m)*64 + col] + siluf(ap[r] + bn2[col]);
      }
    }
  }
}

extern "C" void kernel_launch(void* const* d_in, const int* in_sizes, int n_in,
                              void* d_out, int out_size, void* d_ws, size_t ws_size,
                              hipStream_t stream) {
  const float* h   = (const float*)d_in[0];
  const float* x   = (const float*)d_in[1];
  const float* We1 = (const float*)d_in[2];
  const float* be1 = (const float*)d_in[3];
  const float* We2 = (const float*)d_in[4];
  const float* be2 = (const float*)d_in[5];
  const float* Wa  = (const float*)d_in[6];
  const float* ba  = (const float*)d_in[7];
  const float* Wn1 = (const float*)d_in[8];
  const float* bn1 = (const float*)d_in[9];
  const float* Wn2 = (const float*)d_in[10];
  const float* bn2 = (const float*)d_in[11];
  const float* Wp1 = (const float*)d_in[12];
  const float* bp1 = (const float*)d_in[13];
  const float* Wp2 = (const float*)d_in[14];
  const float* bp2 = (const float*)d_in[15];
  const float* Wv  = (const float*)d_in[16];
  const float* lg  = (const float*)d_in[17];

  char* ws = (char*)d_ws;
  size_t oP    = 0;                                              // 640*64 f32
  size_t oQ    = oP    + (size_t)N_ATOMS*64*4;
  size_t oHE   = oQ    + (size_t)N_ATOMS*64*4;                   // N*N*64 f16 (~52 MB)
  size_t oS    = oHE   + (size_t)N_ATOMS*N_ATOMS*64*2;           // N*N*4 f32 (~6.5 MB)
  size_t oPart = oS    + (size_t)N_ATOMS*N_ATOMS*HEADS*4;        // 640*4*NCHUNK*6 f32
  size_t oM    = oPart + (size_t)N_ATOMS*HEADS*NCHUNK*6*4;
  size_t oD    = oM    + (size_t)N_ATOMS*HEADS*4;
  size_t oHA   = oD    + (size_t)N_ATOMS*HEADS*4;                // 640*256 f32
  size_t oCB   = oHA   + (size_t)N_ATOMS*256*4;                  // 640*64*3 f32

  float*    P    = (float*)(ws + oP);
  float*    Q    = (float*)(ws + oQ);
  _Float16* he   = (_Float16*)(ws + oHE);
  float*    S    = (float*)(ws + oS);
  float*    part = (float*)(ws + oPart);
  float*    M    = (float*)(ws + oM);
  float*    D    = (float*)(ws + oD);
  float*    HA   = (float*)(ws + oHA);
  float*    CB   = (float*)(ws + oCB);
  float*    outp = (float*)d_out;

  k0_precompute<<<dim3(N_ATOMS/16), dim3(32), 0, stream>>>(h, We1, be1, P, Q);
  k1_edge<<<dim3(NCHUNK, N_ATOMS/16), dim3(256), 0, stream>>>(
      x, We1, We2, be2, Wa, ba, lg, P, Q, he, S, part);
  k1b_reduce<<<dim3((N_ATOMS*HEADS + 255)/256), dim3(256), 0, stream>>>(part, M, D);
  k2_aggregate<<<dim3(N_ATOMS/8), dim3(256), 0, stream>>>(x, he, S, M, D, HA, CB);
  k3_node<<<dim3(N_ATOMS/16), dim3(32), 0, stream>>>(
      h, x, Wn1, bn1, Wn2, bn2, Wp1, bp1, Wp2, bp2, Wv, HA, CB, outp);
}
